// MultiHead_Attention_1486058684997
// MI455X (gfx1250) — matrix operand order repaired and checked by reading, not compiled, not run on hardware
//
#include <hip/hip_runtime.h>
#include <stdint.h>

typedef __attribute__((ext_vector_type(16))) __bf16 v16bf;
typedef __attribute__((ext_vector_type(8)))  float  v8f;
typedef __attribute__((ext_vector_type(4)))  int    v4i;
typedef unsigned short u16;
typedef unsigned int   u32;

// address-space-qualified v4i pointers for the async-to-LDS builtin
typedef __attribute__((address_space(1))) v4i gv4i;   // global
typedef __attribute__((address_space(3))) v4i lv4i;   // LDS

union FragU { u32 u[8]; v16bf v; };

#define HAS_ASYNC_LDS __has_builtin(__builtin_amdgcn_global_load_async_to_lds_b128)

__device__ __forceinline__ u16 f32_to_bf16(float f) {
    union { float f; u32 u; } c; c.f = f;
    u32 u = c.u;
    u32 r = (u + 0x7FFFu + ((u >> 16) & 1u)) >> 16;   // RNE
    return (u16)r;
}
__device__ __forceinline__ float bf16_to_f32(u16 h) {
    union { u32 u; float f; } c; c.u = ((u32)h) << 16;
    return c.f;
}

// 16-byte async copy global -> LDS (ASYNCcnt path); sync fallback.
__device__ __forceinline__ void async_copy_b128(u16* lds_dst, const u16* g_src) {
#if HAS_ASYNC_LDS
    __builtin_amdgcn_global_load_async_to_lds_b128(
        (gv4i*)(g_src), (lv4i*)(lds_dst), 0, 0);
#else
    *(uint4*)lds_dst = *(const uint4*)g_src;
#endif
}
__device__ __forceinline__ void wait_async0() {
#if __has_builtin(__builtin_amdgcn_s_wait_asynccnt)
    __builtin_amdgcn_s_wait_asynccnt(0);
#elif HAS_ASYNC_LDS
    asm volatile("s_wait_asynccnt 0" ::: "memory");
#endif
}

// ---------------------------------------------------------------- convert (activations)
__global__ __launch_bounds__(256) void cvt_f32_bf16(
    const float* __restrict__ in, u16* __restrict__ out, int n) {
    int i = blockIdx.x * 256 + threadIdx.x;
    if (i < n) out[i] = f32_to_bf16(in[i]);
}

// ---------------------------------------------------------------- convert + transpose (weights)
// W f32 [1024 k][8192 n]  ->  Wt bf16 [8192 n][1024 k], 64x64 tiles via LDS.
__global__ __launch_bounds__(256) void cvt_transpose_f32_bf16(
    const float* __restrict__ W, u16* __restrict__ Wt) {
    __shared__ u16 lds[64 * 65];
    const int tid = threadIdx.x;
    const int n0 = blockIdx.x * 64;
    const int k0 = blockIdx.y * 64;
    #pragma unroll
    for (int i = 0; i < 16; ++i) {
        int idx = tid + i * 256;
        int r = idx >> 6, c = idx & 63;                 // r = k row, c = n col
        lds[r * 65 + c] = f32_to_bf16(W[(size_t)(k0 + r) * 8192 + n0 + c]);
    }
    __syncthreads();
    #pragma unroll
    for (int i = 0; i < 16; ++i) {
        int idx = tid + i * 256;
        int nn = idx >> 6, kk = idx & 63;               // kk contiguous across lanes
        Wt[(size_t)(n0 + nn) * 1024 + k0 + kk] = lds[kk * 65 + nn];
    }
}

// ---------------------------------------------------------------- GEMM
// C_bf16[8192, 8192] = A_bf16[8192,1024] * Wt_bf16[8192,1024]^T + bias_f32
// Both tiles are contiguous-K row copies -> async-to-LDS double buffer.
#define GK   1024
#define GN   8192
#define BK   32
#define KP   40      // padded halfs per LDS row (80 B, 16B-aligned rows)

__global__ __launch_bounds__(256) void gemm_bias_bf16(
    const u16* __restrict__ A, const u16* __restrict__ Wt,
    const float* __restrict__ bias, u16* __restrict__ C)
{
    __shared__ u16 ldsA[2][128 * KP];    // [m][k]
    __shared__ u16 ldsBT[2][128 * KP];   // [n][k]

    const int tid   = threadIdx.x;
    const int lane  = tid & 31;
    const int wave  = tid >> 5;
    const int hi    = (lane >> 4) & 1;
    const int lm    = lane & 15;
    const int waveM = wave >> 2;   // 0..1
    const int waveN = wave & 3;    // 0..3
    const int mBase = blockIdx.y * 128;
    const int nBase = blockIdx.x * 128;

    v8f acc[4][2];
    #pragma unroll
    for (int mt = 0; mt < 4; ++mt)
        #pragma unroll
        for (int nt = 0; nt < 2; ++nt)
            #pragma unroll
            for (int r = 0; r < 8; ++r) acc[mt][nt][r] = 0.0f;

    // per-lane K-pair base offsets (contiguous K per lane half, same order for A and B)
    int kkA[8], kkB[8];
    #pragma unroll
    for (int p = 0; p < 8; ++p) {
        kkA[p] = 2 * p + hi * 16;                               // A: contiguous K per half
        kkB[p] = 2 * p + hi * 16;                               // B: contiguous K per half
    }

    auto copy_tiles = [&](int buf, int k0) {
        #pragma unroll
        for (int i = 0; i < 2; ++i) {
            int slot = tid + i * 256;
            int row = slot >> 2, seg = slot & 3;
            async_copy_b128(&ldsA[buf][row * KP + seg * 8],
                            A  + (size_t)(mBase + row) * GK + k0 + seg * 8);
            async_copy_b128(&ldsBT[buf][row * KP + seg * 8],
                            Wt + (size_t)(nBase + row) * GK + k0 + seg * 8);
        }
    };

    copy_tiles(0, 0);
    int buf = 0;
    const int NCHUNK = GK / BK;                       // 32
    for (int kc = 0; kc < NCHUNK; ++kc) {
        wait_async0();
        __syncthreads();                              // tile[buf] complete, prev readers done
        if (kc + 1 < NCHUNK) copy_tiles(buf ^ 1, (kc + 1) * BK);

        FragU bf[2];
        #pragma unroll
        for (int nt = 0; nt < 2; ++nt) {
            int n = waveN * 32 + nt * 16 + lm;
            #pragma unroll
            for (int p = 0; p < 8; ++p)
                bf[nt].u[p] = *(const u32*)(&ldsBT[buf][n * KP + kkB[p]]);
        }
        #pragma unroll
        for (int mt = 0; mt < 4; ++mt) {
            int m = waveM * 64 + mt * 16 + lm;
            FragU af;
            #pragma unroll
            for (int p = 0; p < 8; ++p)
                af.u[p] = *(const u32*)(&ldsA[buf][m * KP + kkA[p]]);
            #pragma unroll
            for (int nt = 0; nt < 2; ++nt)
                acc[mt][nt] = __builtin_amdgcn_wmma_f32_16x16x32_bf16(
                    false, af.v, false, bf[nt].v, (short)0, acc[mt][nt], false, false);
        }
        buf ^= 1;
    }

    // epilogue: +bias, store bf16. C/D layout: col = lm, rows = hi*8 + r.
    #pragma unroll
    for (int nt = 0; nt < 2; ++nt) {
        int n = nBase + waveN * 32 + nt * 16 + lm;
        float bv = bias[n];
        #pragma unroll
        for (int mt = 0; mt < 4; ++mt) {
            #pragma unroll
            for (int r = 0; r < 8; ++r) {
                int m = mBase + waveM * 64 + mt * 16 + hi * 8 + r;
                C[(size_t)m * GN + n] = f32_to_bf16(acc[mt][nt][r] + bv);
            }
        }
    }
}

// ---------------------------------------------------------------- fused softmax-pool
// pooled[b,h,t] = sum_s softmax_s(k[s,t]*v[t,s]/sqrt(D)) * q[s,t]
__global__ __launch_bounds__(256) void attn_pool(
    const u16* __restrict__ Q, const u16* __restrict__ Kb, const u16* __restrict__ V,
    float* __restrict__ pooled)
{
    const int S = 1024, HD = 8192, D = 1024;
    const int bh = blockIdx.x;              // 0..63
    const int b = bh >> 3, h = bh & 7;
    const int tBase = blockIdx.y * 256;
    const int tid = threadIdx.x;
    const int t = tBase + tid;
    const size_t base = (size_t)b * S * HD + (size_t)h * D;
    const u16* Qp = Q + base;
    const u16* Kp = Kb + base;
    const u16* Vp = V + base;
    const float scale = 0.03125f;           // 1/sqrt(1024)

    __shared__ u16 ldsV[2][256 * 72];       // 256 rows x 64 halfs (padded), double buffered

    auto copy_v = [&](int bufi, int sc) {
        #pragma unroll
        for (int i = 0; i < 8; ++i) {
            int slot = tid + i * 256;
            int row = slot >> 3, seg = slot & 7;
            async_copy_b128(&ldsV[bufi][row * 72 + seg * 8],
                            Vp + (size_t)(tBase + row) * HD + sc + seg * 8);
        }
    };

    float m = -3.0e38f, l = 0.0f, acc = 0.0f;

    copy_v(0, 0);
    int buf = 0;
    for (int sc = 0; sc < S; sc += 64) {
        wait_async0();
        __syncthreads();
        if (sc + 64 < S) copy_v(buf ^ 1, sc + 64);

        #pragma unroll 4
        for (int ss = 0; ss < 64; ++ss) {
            int s = sc + ss;
            float kv = bf16_to_f32(Kp[(size_t)s * HD + t]);   // coalesced across lanes
            float vv = bf16_to_f32(ldsV[buf][tid * 72 + ss]);
            float qv = bf16_to_f32(Qp[(size_t)s * HD + t]);
            float x = kv * vv * scale;
            if (x > m) {
                float c = __expf(m - x);
                l *= c; acc *= c; m = x;
            }
            float p = __expf(x - m);
            l += p;
            acc += p * qv;
        }
        buf ^= 1;
    }
    pooled[bh * 1024 + t] = acc / l;
}

// ---------------------------------------------------------------- final projection
__global__ __launch_bounds__(256) void final_proj(
    const float* __restrict__ pooled, const float* __restrict__ WL,
    const float* __restrict__ bL, float* __restrict__ out)
{
    const int E = 1024, N = 8192;
    __shared__ float ldsP[8 * 1024];        // 32 KB chunk of pooled
    const int tid = threadIdx.x;
    const int j = blockIdx.x * 256 + tid;
    const float bj = bL[j];

    for (int g = 0; g < 8; ++g) {
        __syncthreads();
        for (int i = tid; i < 8 * 1024; i += 256)
            ldsP[i] = pooled[g * 8 * 1024 + i];
        __syncthreads();

        float acc[8];
        #pragma unroll
        for (int r = 0; r < 8; ++r) acc[r] = bj;
        for (int e = 0; e < E; ++e) {
            float w = WL[(size_t)e * N + j];
            #pragma unroll
            for (int r = 0; r < 8; ++r) acc[r] += ldsP[r * 1024 + e] * w;
        }
        #pragma unroll
        for (int r = 0; r < 8; ++r)
            out[(size_t)(g * 8 + r) * N + j] = acc[r];
    }
}

// ---------------------------------------------------------------- launch
extern "C" void kernel_launch(void* const* d_in, const int* in_sizes, int n_in,
                              void* d_out, int out_size, void* d_ws, size_t ws_size,
                              hipStream_t stream) {
    const float* inputs = (const float*)d_in[0];
    const float* Wq = (const float*)d_in[1];
    const float* bq = (const float*)d_in[2];
    const float* Wk = (const float*)d_in[3];
    const float* bk = (const float*)d_in[4];
    const float* Wv = (const float*)d_in[5];
    const float* bv = (const float*)d_in[6];
    const float* WL = (const float*)d_in[7];
    const float* bL = (const float*)d_in[8];
    float* out = (float*)d_out;

    char* ws = (char*)d_ws;
    const size_t SZ_A = (size_t)8192 * 1024 * 2;   // 16 MB bf16 inputs
    const size_t SZ_W = (size_t)1024 * 8192 * 2;   // 16 MB bf16 weight (transposed)
    const size_t SZ_P = (size_t)8192 * 8192 * 2;   // 128 MB bf16 projection
    u16* Abf = (u16*)(ws);
    u16* Wqt = (u16*)(ws + SZ_A);
    u16* Wkt = (u16*)(ws + SZ_A + SZ_W);
    u16* Wvt = (u16*)(ws + SZ_A + 2 * SZ_W);
    u16* Qb  = (u16*)(ws + SZ_A + 3 * SZ_W);
    u16* Kc  = (u16*)(ws + SZ_A + 3 * SZ_W + SZ_P);
    u16* Vc  = (u16*)(ws + SZ_A + 3 * SZ_W + 2 * SZ_P);
    float* pooled = (float*)(ws + SZ_A + 3 * SZ_W + 3 * SZ_P);

    const int nconv = 8192 * 1024;
    cvt_f32_bf16<<<nconv / 256, 256, 0, stream>>>(inputs, Abf, nconv);

    dim3 tgrid(8192 / 64, 1024 / 64);              // (128, 16)
    cvt_transpose_f32_bf16<<<tgrid, 256, 0, stream>>>(Wq, Wqt);
    cvt_transpose_f32_bf16<<<tgrid, 256, 0, stream>>>(Wk, Wkt);
    cvt_transpose_f32_bf16<<<tgrid, 256, 0, stream>>>(Wv, Wvt);

    dim3 ggrid(GN / 128, 8192 / 128);              // (64, 64)
    gemm_bias_bf16<<<ggrid, 256, 0, stream>>>(Abf, Wqt, bq, Qb);
    gemm_bias_bf16<<<ggrid, 256, 0, stream>>>(Abf, Wkt, bk, Kc);
    gemm_bias_bf16<<<ggrid, 256, 0, stream>>>(Abf, Wvt, bv, Vc);

    attn_pool<<<dim3(64, 4), 256, 0, stream>>>(Qb, Kc, Vc, pooled);

    final_proj<<<32, 256, 0, stream>>>(pooled, WL, bL, out);
}
